// GroupedGemmRunner_6365141532677
// MI455X (gfx1250) — matrix operand order repaired and checked by reading, not compiled, not run on hardware
//
#include <hip/hip_runtime.h>
#include <math.h>

typedef __attribute__((ext_vector_type(16))) __bf16 v16bf;
typedef __attribute__((ext_vector_type(8)))  __bf16 v8bf;
typedef __attribute__((ext_vector_type(8)))  float  v8f;
typedef __attribute__((ext_vector_type(4)))  float  f32x4;
typedef __attribute__((ext_vector_type(2)))  unsigned int u32x2;

#define BM 128
#define BN 128
#define KS 32          // K elements per LDS stage (one WMMA k-step)
#define LDA 40         // KS + 8 pad (bf16 elements); row stride 80B, 16B aligned
#define THREADS 256
#define MAXE 64

// Hardware packed fp32->bf16 (RNE): one VALU op per float pair.
__device__ __forceinline__ unsigned int pack2bf(float x, float y) {
    unsigned int r;
    asm("v_cvt_pk_bf16_f32 %0, %1, %2" : "=v"(r) : "v"(x), "v"(y));
    return r;   // [15:0]=bf16(x), [31:16]=bf16(y)
}

// One pipeline stage. CUR is a compile-time literal 0/1;
// K0 the K-offset of the *next* stage.  Static buffer indices
// keep all LDS addresses as constant offsets (no per-iter select/mad).
#define GG_PREFETCH(K0)                                                     \
    {                                                                       \
        _Pragma("unroll")                                                   \
        for (int i = 0; i < 4; ++i) {                                       \
            ra[i] = *(const f32x4*)(gA[i] + (K0));                          \
            rb[i] = *(const f32x4*)(gB[i] + (K0));                          \
        }                                                                   \
    }

#define GG_COMPUTE(CUR)                                                     \
    {                                                                       \
        v16bf afr[2], bfr[4];                                               \
        _Pragma("unroll")                                                   \
        for (int mi = 0; mi < 2; ++mi) {                                    \
            union { v16bf v; v8bf h[2]; } u;                                \
            u.h[0] = *(const v8bf*)&sA[CUR][frA[mi]];                       \
            u.h[1] = *(const v8bf*)&sA[CUR][frA[mi] + 16];                  \
            afr[mi] = u.v;                                                  \
        }                                                                   \
        _Pragma("unroll")                                                   \
        for (int ni = 0; ni < 4; ++ni) {                                    \
            union { v16bf v; v8bf h[2]; } u;                                \
            u.h[0] = *(const v8bf*)&sB[CUR][frB[ni]];                       \
            u.h[1] = *(const v8bf*)&sB[CUR][frB[ni] + 16];                  \
            bfr[ni] = u.v;                                                  \
        }                                                                   \
        _Pragma("unroll")                                                   \
        for (int mi = 0; mi < 2; ++mi)                                      \
            _Pragma("unroll")                                               \
            for (int ni = 0; ni < 4; ++ni)                                  \
                acc[mi][ni] = __builtin_amdgcn_wmma_f32_16x16x32_bf16(      \
                    false, afr[mi], false, bfr[ni],                         \
                    (short)0, acc[mi][ni], false, false);                   \
    }

#define GG_CVT_STORE(NXT)                                                   \
    {                                                                       \
        _Pragma("unroll")                                                   \
        for (int i = 0; i < 4; ++i) {                                       \
            u32x2 pa, pb;                                                   \
            pa[0] = pack2bf(ra[i][0], ra[i][1]);                            \
            pa[1] = pack2bf(ra[i][2], ra[i][3]);                            \
            pb[0] = pack2bf(rb[i][0], rb[i][1]);                            \
            pb[1] = pack2bf(rb[i][2], rb[i][3]);                            \
            *(u32x2*)&sA[NXT][stOff + i * (32 * LDA)] = pa;                 \
            *(u32x2*)&sB[NXT][stOff + i * (32 * LDA)] = pb;                 \
        }                                                                   \
    }

__global__ __launch_bounds__(THREADS)
void grouped_gemm_bf16_wmma(const float* __restrict__ A,
                            const float* __restrict__ B,
                            float* __restrict__ C,
                            const long long* __restrict__ segptr,
                            const long long* __restrict__ widx,
                            int E, int T, int N, int K)
{
    __shared__ unsigned short sA[2][BM * LDA];
    __shared__ unsigned short sB[2][BN * LDA];
    __shared__ long long sSeg[MAXE + 1];
    __shared__ int sW[MAXE];

    const int tid = threadIdx.x;
    const int m0 = blockIdx.y * BM;
    const int n0 = blockIdx.x * BN;

    for (int i = tid; i < E + 1; i += THREADS) sSeg[i] = segptr[i];
    for (int i = tid; i < E; i += THREADS)     sW[i]  = (int)widx[i];
    __syncthreads();

    const int lane = tid & 31;
    const int wave = tid >> 5;
    const int wm   = wave >> 1;          // 0..3  (rows of 32)
    const int wn   = wave & 1;           // 0..1  (cols of 64)
    const int l16  = lane & 15;
    const int lhi  = lane >> 4;          // 0 or 1

    // global->LDS loader mapping: 4 float4 per matrix per thread per stage
    const int lr = tid >> 3;             // row base 0..31 (then +32,+64,+96)
    const int lc = (tid & 7) * 4;        // float (==bf16) column within KS

    // Hoisted loop-invariant global row pointers (constant-indexed arrays only)
    const float* gA[4];
    int rowB[4];
    #pragma unroll
    for (int i = 0; i < 4; ++i) {
        int rA = m0 + lr + i * 32; if (rA >= T) rA = T - 1;
        int rB = n0 + lr + i * 32; if (rB >= N) rB = N - 1;
        rowB[i] = rB;
        gA[i] = A + (size_t)rA * K + lc;
    }
    const int stOff = lr * LDA + lc;     // LDS store base (element index)
    // fragment read base offsets (element offsets within a buffer)
    int frA[2], frB[4];
    #pragma unroll
    for (int mi = 0; mi < 2; ++mi)
        frA[mi] = (wm * 32 + mi * 16 + l16) * LDA + lhi * 8;
    #pragma unroll
    for (int ni = 0; ni < 4; ++ni)
        frB[ni] = (wn * 64 + ni * 16 + l16) * LDA + lhi * 8;

    const int NKS = K / KS;

    for (int e = 0; e < E; ++e) {
        const int lo = max(m0, (int)sSeg[e]);
        const int hi = min(m0 + BM, (int)sSeg[e + 1]);
        if (lo >= hi) continue;

        const float* __restrict__ W = B + (size_t)sW[e] * (size_t)N * (size_t)K;
        const float* gB[4];
        #pragma unroll
        for (int i = 0; i < 4; ++i)
            gB[i] = W + (size_t)rowB[i] * K + lc;

        v8f acc[2][4];
        #pragma unroll
        for (int mi = 0; mi < 2; ++mi)
            #pragma unroll
            for (int ni = 0; ni < 4; ++ni)
                acc[mi][ni] = (v8f){0.f,0.f,0.f,0.f,0.f,0.f,0.f,0.f};

        f32x4 ra[4], rb[4];

        // ---- prologue: stage 0 -> buf 0 ----
        GG_PREFETCH(0);
        GG_CVT_STORE(0);
        __syncthreads();

        // ---- pair-unrolled pipeline: static buffer indices 0->1->0 ----
        int ks = 0;
        for (; ks + 2 < NKS; ks += 2) {
            GG_PREFETCH((ks + 1) * KS);
            GG_COMPUTE(0);
            GG_CVT_STORE(1);
            __syncthreads();

            GG_PREFETCH((ks + 2) * KS);
            GG_COMPUTE(1);
            GG_CVT_STORE(0);
            __syncthreads();
        }
        // ---- peeled tail: 1 or 2 stages remain ----
        if (NKS - ks == 2) {
            GG_PREFETCH((ks + 1) * KS);
            GG_COMPUTE(0);
            GG_CVT_STORE(1);
            __syncthreads();
            GG_COMPUTE(1);
            __syncthreads();
        } else {
            GG_COMPUTE(0);
            __syncthreads();
        }

        // ---- store: rows masked to this expert's segment ----
        #pragma unroll
        for (int mi = 0; mi < 2; ++mi) {
            #pragma unroll
            for (int ni = 0; ni < 4; ++ni) {
                const int n = n0 + wn * 64 + ni * 16 + l16;
                #pragma unroll
                for (int v = 0; v < 8; ++v) {
                    const int m = m0 + wm * 32 + mi * 16 + lhi * 8 + v;
                    if (m >= lo && m < hi && n < N)
                        C[(size_t)m * N + n] = acc[mi][ni][v];
                }
            }
        }
    }
}

extern "C" void kernel_launch(void* const* d_in, const int* in_sizes, int n_in,
                              void* d_out, int out_size, void* d_ws, size_t ws_size,
                              hipStream_t stream) {
    const float* a        = (const float*)d_in[0];
    const float* b        = (const float*)d_in[1];
    const long long* seg  = (const long long*)d_in[3];
    const long long* widx = (const long long*)d_in[4];
    float* out            = (float*)d_out;

    const int E = in_sizes[4];                        // len(weight_indices)
    const long long s0  = in_sizes[0];                // T*K
    const long long sNK = in_sizes[1] / (long long)E; // N*K
    // K^2 = (T*K)*(N*K)/(T*N)
    const double kd = sqrt((double)s0 * (double)sNK / (double)out_size);
    const int K = (int)(kd + 0.5);
    const int T = (int)(s0 / K);
    const int N = (int)(sNK / K);

    dim3 grid((N + BN - 1) / BN, (T + BM - 1) / BM);
    grouped_gemm_bf16_wmma<<<grid, dim3(THREADS), 0, stream>>>(
        a, b, out, seg, widx, E, T, N, K);
}
